// NdpCnn_49014166782545
// MI455X (gfx1250) — compile-verified, matrix operand order checked by reading, not run on hardware
//
#include <hip/hip_runtime.h>

// ---------------------------------------------------------------------------
// Types for CDNA5 WMMA (wave32)
// ---------------------------------------------------------------------------
typedef __attribute__((ext_vector_type(16))) _Float16 v16h;
typedef __attribute__((ext_vector_type(8)))  float    v8f;

#define N_RBF   30
#define T_STEPS 300
#define DT      (1.0f / 300.0f)
#define A_Z     25.0f
#define B_Z     6.25f

// Fragment loading convention (K-permutation trick):
// WMMA reduces over K; since we control BOTH operands' memory layouts, any
// bijective remap memory-k -> hardware-k-slot is an identity transform as long
// as A and B use the same remap. We choose: lane half h = (lane>>4) reads the
// contiguous 16xf16 range [k0 + 16h, k0 + 16h + 16). One aligned 32-byte v16h
// load per fragment (2x b128 into adjacent VGPRs), no shuffles, no gathers.

// ---------------------------------------------------------------------------
// Kernel 1: conv1 (1->20ch, 5x5, VALID) + ReLU + 2x2 maxpool, fused, fp32.
// ---------------------------------------------------------------------------
__global__ void conv1_pool_kernel(const float* __restrict__ in,
                                  const float* __restrict__ w,
                                  const float* __restrict__ bias,
                                  _Float16* __restrict__ out, int total) {
    __shared__ float sw[20 * 25];
    __shared__ float sb[20];
    for (int i = threadIdx.x; i < 500; i += blockDim.x) sw[i] = w[i];
    for (int i = threadIdx.x; i < 20;  i += blockDim.x) sb[i] = bias[i];
    __syncthreads();

    int idx = blockIdx.x * blockDim.x + threadIdx.x;
    if (idx >= total) return;
    int px = idx % 12;
    int t  = idx / 12;
    int py = t % 12;  t /= 12;
    int oc = t % 20;
    int b  = t / 20;

    const float* ip = in + (size_t)b * 784;
    const int y0 = py * 2, x0 = px * 2;
    float patch[36];
#pragma unroll
    for (int i = 0; i < 6; ++i)
#pragma unroll
        for (int j = 0; j < 6; ++j)
            patch[i * 6 + j] = ip[(y0 + i) * 28 + (x0 + j)];

    const float* wk = &sw[oc * 25];
    const float bv = sb[oc];
    float best = 0.0f;  // max of ReLU values is >= 0
#pragma unroll
    for (int dy = 0; dy < 2; ++dy)
#pragma unroll
        for (int dx = 0; dx < 2; ++dx) {
            float acc = bv;
#pragma unroll
            for (int i = 0; i < 5; ++i)
#pragma unroll
                for (int j = 0; j < 5; ++j)
                    acc = fmaf(patch[(dy + i) * 6 + (dx + j)], wk[i * 5 + j], acc);
            best = fmaxf(best, acc);
        }
    out[idx] = (_Float16)best;
}

// ---------------------------------------------------------------------------
// Kernel 2: generic weight pack fp32 (N,Kin) -> zero-padded f16 [Npad][Kpad].
// ---------------------------------------------------------------------------
__global__ void pack_w_kernel(const float* __restrict__ W,
                              _Float16* __restrict__ out,
                              int N, int Kin, int Kpad, int total) {
    int idx = blockIdx.x * blockDim.x + threadIdx.x;
    if (idx >= total) return;
    int n = idx / Kpad;
    int k = idx - n * Kpad;
    out[idx] = (n < N && k < Kin) ? (_Float16)W[n * Kin + k] : (_Float16)0.0f;
}

// ---------------------------------------------------------------------------
// Kernel 3: conv2 as im2col WMMA GEMM + bias + ReLU + 2x2 maxpool.
// M = B*64, N = 50 (pad 64), K = 500 (pad 512).
// Per block (image b, row-pair t): the 16x512 im2col A-tile is materialized in
// LDS by a cooperative fill (division chains run once per element), then the
// hot loop is pure aligned ds_load_b128 (A) + global_load_b128 (B) + v_wmma.
// 2x2 maxpool completes inside the wave (VGPR pairs + shfl_xor 16).
// ---------------------------------------------------------------------------
__global__ void conv2_wmma_kernel(const _Float16* __restrict__ act1,
                                  const _Float16* __restrict__ wf16,
                                  const float* __restrict__ bias,
                                  _Float16* __restrict__ act2) {
    __shared__ __align__(64) _Float16 As[16 * 512];   // 16 KB im2col tile

    const int b   = blockIdx.x >> 2;
    const int t   = blockIdx.x & 3;         // pooled output row py
    const int tid = threadIdx.x;

    const _Float16* img = act1 + (size_t)b * 2880;    // (20,12,12) f16
    // Cooperative im2col fill: row m -> output pos (oy=2t+m/8, ox=m%8).
    for (int i = tid; i < 16 * 512; i += 128) {
        int m = i >> 9;
        int k = i & 511;
        _Float16 v = (_Float16)0.0f;
        if (k < 500) {
            int ic  = k / 25;
            int rem = k - ic * 25;
            int r   = rem / 5;
            int s   = rem - r * 5;
            int oy  = 2 * t + (m >> 3);
            int ox  = m & 7;
            v = img[ic * 144 + (oy + r) * 12 + (ox + s)];
        }
        As[i] = v;
    }
    __syncthreads();

    const int ntile = tid >> 5;             // 0..3 (N tile)
    const int lane  = tid & 31;
    const int m     = lane & 15;
    const int koff  = (lane >> 4) * 16;     // contiguous-K remap (see header)
    const int nB    = ntile * 16;

    const _Float16* arow = As + m * 512 + koff;
    const _Float16* brow = wf16 + (size_t)(nB + m) * 512 + koff;

    v8f c = {};
#pragma unroll 4
    for (int k0 = 0; k0 < 512; k0 += 32) {
        v16h a  = *(const v16h*)(arow + k0);
        v16h bb = *(const v16h*)(brow + k0);
        c = __builtin_amdgcn_wmma_f32_16x16x32_f16(false, a, false, bb,
                                                   (short)0, c, false, false);
    }

    // Epilogue: bias + ReLU + in-wave 2x2 maxpool.
    const int n = nB + m;
    const float bv = (n < 50) ? bias[n] : 0.0f;
    float r[8];
#pragma unroll
    for (int v = 0; v < 8; ++v) {
        float x = c[v] + bv;
        r[v] = x > 0.0f ? x : 0.0f;
    }
#pragma unroll
    for (int px = 0; px < 4; ++px) {
        float q = fmaxf(r[2 * px], r[2 * px + 1]);
        q = fmaxf(q, __shfl_xor(q, 16, 32));
        if (lane < 16 && n < 50) {
            act2[(size_t)b * 800 + n * 16 + t * 4 + px] = (_Float16)q;
        }
    }
}

// ---------------------------------------------------------------------------
// Kernel 4: dense layer via WMMA on pre-packed f16 weights.
// out = relu(in @ Wp^T + bias); pad columns [N, strideOut) written as zeros so
// the next layer's K loop needs no guards. One aligned v16h load per fragment.
// One wave (block of 32) per 16x16 C tile.
// ---------------------------------------------------------------------------
__global__ void fc_wmma_kernel(const _Float16* __restrict__ in,
                               const _Float16* __restrict__ Wp,
                               const float* __restrict__ bias,
                               _Float16* __restrict__ out,
                               int Kpad, int N, int strideOut, int ntiles) {
    const int mtile = blockIdx.x / ntiles;
    const int ntile = blockIdx.x % ntiles;
    const int lane  = threadIdx.x;
    const int m     = lane & 15;
    const int koff  = (lane >> 4) * 16;     // contiguous-K remap (see header)

    const _Float16* arow = in + (size_t)(mtile * 16 + m) * Kpad + koff;
    const _Float16* brow = Wp + (size_t)(ntile * 16 + m) * Kpad + koff;

    v8f c = {};
#pragma unroll 4
    for (int k0 = 0; k0 < Kpad; k0 += 32) {
        v16h a = *(const v16h*)(arow + k0);
        v16h b = *(const v16h*)(brow + k0);
        c = __builtin_amdgcn_wmma_f32_16x16x32_f16(false, a, false, b,
                                                   (short)0, c, false, false);
    }

    const int n  = ntile * 16 + m;
    const float bv = (n < N) ? bias[n] : 0.0f;
#pragma unroll
    for (int v = 0; v < 8; ++v) {
        int row = mtile * 16 + v + ((lane >> 4) * 8);
        float x = c[v] + bv;
        x = (n < N && x > 0.0f) ? x : 0.0f;   // ReLU; pad cols -> 0
        out[(size_t)row * strideOut + n] = (_Float16)x;
    }
}

// ---------------------------------------------------------------------------
// Kernel 5: final linear head in fp32 (x @ last_w^T + b) * 1000 -> DMP params.
// ---------------------------------------------------------------------------
__global__ void last_layer_kernel(const _Float16* __restrict__ in,
                                  const float* __restrict__ W,
                                  const float* __restrict__ bias,
                                  float* __restrict__ params, int total) {
    int idx = blockIdx.x * blockDim.x + threadIdx.x;  // b*32 + n
    if (idx >= total) return;
    int n = idx & 31;
    int b = idx >> 5;
    const _Float16* x = in + (size_t)b * 32;
    const float* wr = W + n * 32;
    float acc = bias[n];
#pragma unroll
    for (int k = 0; k < 32; ++k) acc = fmaf((float)x[k], wr[k], acc);
    params[idx] = acc * 1000.0f;
}

// ---------------------------------------------------------------------------
// Kernel 6: RBF basis table psi(s,n), row sums, canonical xs (batch-invariant).
// ---------------------------------------------------------------------------
__global__ void psi_kernel(float* __restrict__ psi, float* __restrict__ psum,
                           float* __restrict__ xs_arr) {
    int s = blockIdx.x * blockDim.x + threadIdx.x;
    if (s >= T_STEPS) return;
    const float base = 1.0f - DT;                        // (1 - a_x*dt/tau)
    float xs = __powf(base, (float)(s + 1));
    float sum = 0.0f;
#pragma unroll
    for (int n = 0; n < N_RBF; ++n) {
        float cn = __expf(-(float)n * (1.0f / 29.0f));   // exp(-a_x*linspace)
        float hn = 164.31676725154983f / cn;             // 30^1.5 / c / a_x
        float d  = xs - cn;
        float p  = __expf(-hn * d * d);
        psi[s * 32 + n] = p;
        sum += p;
    }
    psum[s]   = sum;
    xs_arr[s] = xs;
}

// ---------------------------------------------------------------------------
// Kernel 7: forcing term fx(s,b) = (psi(s,:)·w(b,:)) * xs * (goal-y0) / psum.
// ---------------------------------------------------------------------------
__global__ void fx_kernel(const float* __restrict__ psi,
                          const float* __restrict__ psum,
                          const float* __restrict__ xs,
                          const float* __restrict__ params,
                          const float* __restrict__ y0,
                          float* __restrict__ fx, int B) {
    __shared__ float sp[32];
    __shared__ float sxs, sps;
    const int s = blockIdx.y;
    const int b = blockIdx.x * blockDim.x + threadIdx.x;
    if (threadIdx.x < N_RBF) sp[threadIdx.x] = psi[s * 32 + threadIdx.x];
    if (threadIdx.x == 30) sxs = xs[s];
    if (threadIdx.x == 31) sps = psum[s];
    __syncthreads();
    if (b >= B) return;
    const float* pp = params + (size_t)b * 32;
    const float goal = pp[0];
    float acc = 0.0f;
#pragma unroll
    for (int n = 0; n < N_RBF; ++n) acc = fmaf(sp[n], pp[1 + n], acc);
    fx[(size_t)s * B + b] = acc * sxs * (goal - y0[b]) / sps;
}

// ---------------------------------------------------------------------------
// Kernel 8: DMP integration scan. One thread per batch element; fx loads are
// fully coalesced (stride B). Output (B, 301).
// ---------------------------------------------------------------------------
__global__ void dmp_scan_kernel(const float* __restrict__ fx,
                                const float* __restrict__ params,
                                const float* __restrict__ y0_in,
                                float* __restrict__ out, int B) {
    int b = blockIdx.x * blockDim.x + threadIdx.x;
    if (b >= B) return;
    const float goal = params[(size_t)b * 32];
    float y = y0_in[b];
    float z = 0.01f;                     // dy0 * tau
    float* ob = out + (size_t)b * (T_STEPS + 1);
    ob[0] = y;
    for (int s = 0; s < T_STEPS; ++s) {
        float f  = fx[(size_t)s * B + b];
        float dz = fmaf(A_Z, fmaf(B_Z, goal - y, -z), f);  // tau = 1
        float dy = z;
        y = fmaf(dy, DT, y);
        z = fmaf(dz, DT, z);
        ob[s + 1] = y;
    }
}

// ---------------------------------------------------------------------------
// Host-side launch
// ---------------------------------------------------------------------------
extern "C" void kernel_launch(void* const* d_in, const int* in_sizes, int n_in,
                              void* d_out, int out_size, void* d_ws, size_t ws_size,
                              hipStream_t stream) {
    const float* input   = (const float*)d_in[0];
    const float* y0      = (const float*)d_in[1];
    const float* conv1_w = (const float*)d_in[2];
    const float* conv1_b = (const float*)d_in[3];
    const float* conv2_w = (const float*)d_in[4];
    const float* conv2_b = (const float*)d_in[5];
    const float* fc1_w   = (const float*)d_in[6];
    const float* fc1_b   = (const float*)d_in[7];
    const float* l1_w    = (const float*)d_in[8];
    const float* l1_b    = (const float*)d_in[9];
    const float* l2_w    = (const float*)d_in[10];
    const float* l2_b    = (const float*)d_in[11];
    const float* l3_w    = (const float*)d_in[12];
    const float* l3_b    = (const float*)d_in[13];
    const float* l4_w    = (const float*)d_in[14];
    const float* l4_b    = (const float*)d_in[15];
    const float* last_w  = (const float*)d_in[16];
    const float* last_b  = (const float*)d_in[17];
    float* out = (float*)d_out;

    const int B = in_sizes[0] / 784;   // 4096

    // Workspace layout
    char* ws = (char*)d_ws;
    size_t off = 0;
    auto alloc = [&](size_t bytes) {
        size_t r = off;
        off += (bytes + 255) & ~(size_t)255;
        return r;
    };
    size_t o_act1   = alloc((size_t)B * 2880 * 2);    // conv1 out f16 (B,20,12,12)
    size_t o_wconv2 = alloc((size_t)64 * 512 * 2);    // packed conv2 weights
    size_t o_wfc1   = alloc((size_t)224 * 800 * 2);   // packed fc1  [224][800]
    size_t o_wl1    = alloc((size_t)128 * 224 * 2);   // packed l1   [128][224]
    size_t o_wl2    = alloc((size_t)224 * 128 * 2);   // packed l2   [224][128]
    size_t o_wl3    = alloc((size_t)64 * 224 * 2);    // packed l3   [ 64][224]
    size_t o_wl4    = alloc((size_t)32 * 64 * 2);     // packed l4   [ 32][ 64]
    size_t o_act2   = alloc((size_t)B * 800 * 2);     // conv2 pooled f16 (B,800)
    size_t o_act3   = alloc((size_t)B * 224 * 2);     // fc1 out (stride 224)
    size_t o_act4   = alloc((size_t)B * 128 * 2);     // l1 out (stride 128)
    size_t o_act5   = alloc((size_t)B * 224 * 2);     // l2 out (stride 224)
    size_t o_act6   = alloc((size_t)B * 64 * 2);      // l3 out
    size_t o_act7   = alloc((size_t)B * 32 * 2);      // l4 out
    size_t o_params = alloc((size_t)B * 32 * 4);      // DMP params f32
    size_t o_psi    = alloc((size_t)T_STEPS * 32 * 4);
    size_t o_psum   = alloc((size_t)T_STEPS * 4);
    size_t o_xs     = alloc((size_t)T_STEPS * 4);
    (void)ws_size; (void)n_in; (void)out_size;

    _Float16* act1   = (_Float16*)(ws + o_act1);
    _Float16* wconv2 = (_Float16*)(ws + o_wconv2);
    _Float16* wfc1   = (_Float16*)(ws + o_wfc1);
    _Float16* wl1    = (_Float16*)(ws + o_wl1);
    _Float16* wl2    = (_Float16*)(ws + o_wl2);
    _Float16* wl3    = (_Float16*)(ws + o_wl3);
    _Float16* wl4    = (_Float16*)(ws + o_wl4);
    _Float16* act2   = (_Float16*)(ws + o_act2);
    _Float16* act3   = (_Float16*)(ws + o_act3);
    _Float16* act4   = (_Float16*)(ws + o_act4);
    _Float16* act5   = (_Float16*)(ws + o_act5);
    _Float16* act6   = (_Float16*)(ws + o_act6);
    _Float16* act7   = (_Float16*)(ws + o_act7);
    float* params = (float*)(ws + o_params);
    float* psi    = (float*)(ws + o_psi);
    float* psum   = (float*)(ws + o_psum);
    float* xs     = (float*)(ws + o_xs);
    float* fx     = (float*)(ws + o_act1);   // alias: act1 dead after conv2

    // 0) pack all GEMM weights to zero-padded f16 (tiny, one-shot per call)
    pack_w_kernel<<<(64 * 512 + 255) / 256, 256, 0, stream>>>(conv2_w, wconv2, 50, 500, 512, 64 * 512);
    pack_w_kernel<<<(224 * 800 + 255) / 256, 256, 0, stream>>>(fc1_w, wfc1, 200, 800, 800, 224 * 800);
    pack_w_kernel<<<(128 * 224 + 255) / 256, 256, 0, stream>>>(l1_w, wl1, 100, 200, 224, 128 * 224);
    pack_w_kernel<<<(224 * 128 + 255) / 256, 256, 0, stream>>>(l2_w, wl2, 200, 100, 128, 224 * 128);
    pack_w_kernel<<<(64 * 224 + 255) / 256, 256, 0, stream>>>(l3_w, wl3, 64, 200, 224, 64 * 224);
    pack_w_kernel<<<(32 * 64 + 255) / 256, 256, 0, stream>>>(l4_w, wl4, 32, 64, 64, 32 * 64);

    // 1) conv1 + relu + pool
    {
        int total = B * 20 * 144;
        conv1_pool_kernel<<<(total + 255) / 256, 256, 0, stream>>>(
            input, conv1_w, conv1_b, act1, total);
    }
    // 2) conv2 implicit-GEMM WMMA + relu + pool
    conv2_wmma_kernel<<<B * 4, 128, 0, stream>>>(act1, wconv2, conv2_b, act2);
    // 3) FC chain (WMMA, one wave per tile)
    const int Mt = B / 16;
    fc_wmma_kernel<<<Mt * 14, 32, 0, stream>>>(act2, wfc1, fc1_b, act3, 800, 200, 224, 14);
    fc_wmma_kernel<<<Mt * 8,  32, 0, stream>>>(act3, wl1,  l1_b,  act4, 224, 100, 128, 8);
    fc_wmma_kernel<<<Mt * 14, 32, 0, stream>>>(act4, wl2,  l2_b,  act5, 128, 200, 224, 14);
    fc_wmma_kernel<<<Mt * 4,  32, 0, stream>>>(act5, wl3,  l3_b,  act6, 224, 64,  64,  4);
    fc_wmma_kernel<<<Mt * 2,  32, 0, stream>>>(act6, wl4,  l4_b,  act7, 64,  32,  32,  2);
    // 4) fp32 head -> DMP params (goal, w[30]) scaled x1000
    last_layer_kernel<<<(B * 32 + 255) / 256, 256, 0, stream>>>(
        act7, last_w, last_b, params, B * 32);
    // 5) RBF basis + canonical system (batch-invariant)
    psi_kernel<<<2, 256, 0, stream>>>(psi, psum, xs);
    // 6) forcing term fx (S,B)
    {
        dim3 g(B / 256, T_STEPS);
        fx_kernel<<<g, 256, 0, stream>>>(psi, psum, xs, params, y0, fx, B);
    }
    // 7) ODE scan -> output (B, 301)
    dmp_scan_kernel<<<(B + 255) / 256, 256, 0, stream>>>(fx, params, y0, out, B);
}
